// ScaledDotProductAttention_73675868995933
// MI455X (gfx1250) — matrix operand order repaired and checked by reading, not compiled, not run on hardware
//
#include <hip/hip_runtime.h>
#include <hip/hip_bf16.h>

// CDNA5 (gfx1250, wave32) scaled-dot-product attention, B=4,H=16,S=2048,D=64.
// Outputs: context [B,H,S,D] then attn_score [B,H,S,S], concatenated in d_out.
// v2: base-2 softmax (v_exp_f32/v_log_f32 are exp2/log2 on AMD); the 1/sqrt(d)
// scale and log2(e) are folded into the Q->f16 conversion so the WMMA score
// output is already in the log2 domain. Cuts ~3 VALU ops per score element in
// both passes of this bandwidth/VALU-balanced kernel.

typedef __attribute__((ext_vector_type(16))) _Float16     v16h;
typedef __attribute__((ext_vector_type(8)))  _Float16     h8;
typedef __attribute__((ext_vector_type(4)))  _Float16     h4;
typedef __attribute__((ext_vector_type(8)))  float        v8f;
typedef __attribute__((ext_vector_type(4)))  float        f4;
typedef __attribute__((ext_vector_type(4)))  unsigned int u32x4;

#define SEQ   2048
#define DK    64
#define TQ    128      // query rows per workgroup
#define CK    128      // key chunk per LDS stage
#define NCH   (SEQ / CK)
#define LDQ   72       // padded LDS row stride (halves) for Q/K tiles
#define LDVT  136      // padded LDS row stride (halves) for V^T (64 x 128)
#define LDP   40       // padded LDS row stride (halves) for per-wave P tile

#define QSCALE   0.18033688011112f    // (1/8) * log2(e)
#define ZMASKED  (-1.44269504e9f)     // -1e9 * log2(e)

// ---- WMMA operand loaders (wave32, 16-bit layouts per CDNA5 ISA 7.12.2) ----
static __device__ __forceinline__ v16h load_a16(const _Float16* T, int ld,
                                                int row0, int col0, int l, int hi) {
  const h8* p0 = reinterpret_cast<const h8*>(T + (size_t)(row0 + l) * ld + col0 + hi * 8);
  const h8* p1 = reinterpret_cast<const h8*>(T + (size_t)(row0 + l) * ld + col0 + 16 + hi * 8);
  h8 x = *p0, y = *p1;
  return __builtin_shufflevector(x, y, 0,1,2,3,4,5,6,7,8,9,10,11,12,13,14,15);
}
static __device__ __forceinline__ v16h load_b16(const _Float16* T, int ld,
                                                int row0, int col0, int l, int hi) {
  const h8* p0 = reinterpret_cast<const h8*>(T + (size_t)(row0 + l) * ld + col0 + hi * 8);
  h8 x = p0[0], y = p0[2];
  return __builtin_shufflevector(x, y, 0,1,2,3,4,5,6,7,8,9,10,11,12,13,14,15);
}

__global__ __launch_bounds__(256) void sdpa_fused_kernel(
    const float* __restrict__ Qg, const float* __restrict__ Kg,
    const float* __restrict__ Vg, const unsigned char* __restrict__ Mg,
    float* __restrict__ ctxOut, float* __restrict__ attnOut) {

  __shared__ __align__(16) _Float16     Qh[TQ * LDQ];     // 18 KB
  __shared__ __align__(16) _Float16     Kh[CK * LDQ];     // 18 KB
  __shared__ __align__(16) _Float16     Vt[DK * LDVT];    // 17 KB  (V^T: [d][k])
  __shared__ __align__(16) unsigned int MaskB[TQ * 64];   // 32 KB  (bit = masked)
  __shared__ __align__(16) _Float16     P16[8 * 16 * LDP];// 10 KB  (per-wave P)

  const int tid  = threadIdx.x;
  const int w    = tid >> 5;          // wave id 0..7 -> query rows w*16..w*16+15
  const int lane = tid & 31;
  const int l    = lane & 15;
  const int hi   = lane >> 4;

  const int bh = blockIdx.x >> 4;
  const int qb = blockIdx.x & 15;

  const size_t qkvBase = (size_t)bh * SEQ * DK;
  const float* Qp = Qg + qkvBase + (size_t)qb * TQ * DK;
  const float* Kp = Kg + qkvBase;
  const float* Vp = Vg + qkvBase;
  const size_t mBase = ((size_t)bh * SEQ + (size_t)qb * TQ) * SEQ;  // mask bytes
  const size_t aBase = mBase;                                        // attn floats
  const size_t cBase = ((size_t)bh * SEQ + (size_t)qb * TQ) * DK;    // ctx floats

  // ---- stage Q tile, pre-scaled by (1/sqrt(d))*log2(e), f32 -> f16 ----
  {
    const f4* Q4 = reinterpret_cast<const f4*>(Qp);
#pragma unroll
    for (int i = 0; i < 8; ++i) {
      int idx = tid + i * 256;
      int row = idx >> 4, g = idx & 15;
      f4 q = Q4[idx];
      h4 hq; hq[0] = (_Float16)(q[0] * QSCALE); hq[1] = (_Float16)(q[1] * QSCALE);
      hq[2] = (_Float16)(q[2] * QSCALE); hq[3] = (_Float16)(q[3] * QSCALE);
      *reinterpret_cast<h4*>(&Qh[(size_t)row * LDQ + g * 4]) = hq;
    }
  }
  __syncthreads();

  const v16h a0 = load_a16(Qh, LDQ, w * 16, 0, l, hi);
  const v16h a1 = load_a16(Qh, LDQ, w * 16, 32, l, hi);

  float mreg[8], lreg[8];                 // running log2-max / sum of 2^(z-m)
#pragma unroll
  for (int v = 0; v < 8; ++v) { mreg[v] = -3.0e38f; lreg[v] = 0.0f; }

  // ================= Phase 1: softmax stats (base-2 online) =================
  for (int c = 0; c < NCH; ++c) {
    __syncthreads();
    {
      const f4* K4 = reinterpret_cast<const f4*>(Kp) + (size_t)c * (CK * DK / 4);
#pragma unroll
      for (int i = 0; i < 8; ++i) {
        int idx = tid + i * 256;
        int row = idx >> 4, g = idx & 15;
        f4 k = K4[idx];
        h4 hk; hk[0] = (_Float16)k[0]; hk[1] = (_Float16)k[1];
        hk[2] = (_Float16)k[2]; hk[3] = (_Float16)k[3];
        *reinterpret_cast<h4*>(&Kh[(size_t)row * LDQ + g * 4]) = hk;
      }
    }
    {   // pack 128x128 mask bools -> bits (kept resident for phase 2)
#pragma unroll
      for (int i = 0; i < 4; ++i) {
        int ou = tid + i * 256;
        int row = ou >> 3, grp = ou & 7;
        const u32x4* mp = reinterpret_cast<const u32x4*>(
            Mg + mBase + (size_t)row * SEQ + (size_t)c * CK + grp * 16);
        u32x4 mv = __builtin_nontemporal_load(mp);
        unsigned bits = 0;
#pragma unroll
        for (int ww = 0; ww < 4; ++ww) {
          unsigned word = mv[ww];
#pragma unroll
          for (int b = 0; b < 4; ++b)
            bits |= (((word >> (8 * b)) & 0xFFu) ? 1u : 0u) << (ww * 4 + b);
        }
        reinterpret_cast<unsigned short*>(MaskB)[row * 128 + c * 8 + grp] =
            (unsigned short)bits;
      }
    }
    __syncthreads();

    u32x4 mrA[8];
#pragma unroll
    for (int v = 0; v < 8; ++v)
      mrA[v] = *reinterpret_cast<const u32x4*>(
          MaskB + ((w * 16 + v + 8 * hi) * 64 + c * 4));

#pragma unroll
    for (int nt = 0; nt < 8; ++nt) {
      v16h b0 = load_b16(Kh, LDQ, nt * 16, 0, l, hi);
      v16h b1 = load_b16(Kh, LDQ, nt * 16, 32, l, hi);
      v8f s = {0.f, 0.f, 0.f, 0.f, 0.f, 0.f, 0.f, 0.f};
      s = __builtin_amdgcn_wmma_f32_16x16x32_f16(false, a0, false, b0, (short)0, s, false, false);
      s = __builtin_amdgcn_wmma_f32_16x16x32_f16(false, a1, false, b1, (short)0, s, false, false);
      const int word = nt >> 1;
      const int shift = ((nt & 1) << 4) + l;
#pragma unroll
      for (int v = 0; v < 8; ++v) {
        unsigned bit = (mrA[v][word] >> shift) & 1u;
        float z = bit ? ZMASKED : s[v];               // already log2-domain
        bool zle = z <= mreg[v];
        float nm = zle ? mreg[v] : z;
        float e = __builtin_amdgcn_exp2f((zle ? z : mreg[v]) - nm);
        lreg[v] = zle ? (lreg[v] + e) : (lreg[v] * e + 1.0f);
        mreg[v] = nm;
      }
    }
  }

  // cross-lane merge over the 16 lanes of each row; rowC = -(m + log2(l))
  float rowC[8];
#pragma unroll
  for (int v = 0; v < 8; ++v) {
    float m = mreg[v], ls = lreg[v];
#pragma unroll
    for (int msk = 1; msk < 16; msk <<= 1) {
      float om = __shfl_xor(m, msk, 32);
      float ol = __shfl_xor(ls, msk, 32);
      float nm = fmaxf(m, om);
      ls = ls * __builtin_amdgcn_exp2f(m - nm) +
           ol * __builtin_amdgcn_exp2f(om - nm);
      m = nm;
    }
    rowC[v] = -(m + __builtin_amdgcn_logf(ls));   // v_log_f32 == log2
  }

  // ================= Phase 2: normalize, write attn, accumulate P·V =========
  v8f acc[4];
#pragma unroll
  for (int dt = 0; dt < 4; ++dt) acc[dt] = (v8f){0.f,0.f,0.f,0.f,0.f,0.f,0.f,0.f};

  _Float16* P16w = P16 + (size_t)w * 16 * LDP;

  for (int c = 0; c < NCH; ++c) {
    __syncthreads();
    {
      const f4* K4 = reinterpret_cast<const f4*>(Kp) + (size_t)c * (CK * DK / 4);
      const f4* V4 = reinterpret_cast<const f4*>(Vp) + (size_t)c * (CK * DK / 4);
#pragma unroll
      for (int i = 0; i < 8; ++i) {
        int idx = tid + i * 256;
        int row = idx >> 4, g = idx & 15;
        f4 k = K4[idx];
        h4 hk; hk[0] = (_Float16)k[0]; hk[1] = (_Float16)k[1];
        hk[2] = (_Float16)k[2]; hk[3] = (_Float16)k[3];
        *reinterpret_cast<h4*>(&Kh[(size_t)row * LDQ + g * 4]) = hk;
        f4 vv = V4[idx];
#pragma unroll
        for (int j = 0; j < 4; ++j)
          Vt[(size_t)(g * 4 + j) * LDVT + row] = (_Float16)vv[j];
      }
    }
    __syncthreads();

    u32x4 mrA[8];
#pragma unroll
    for (int v = 0; v < 8; ++v)
      mrA[v] = *reinterpret_cast<const u32x4*>(
          MaskB + ((w * 16 + v + 8 * hi) * 64 + c * 4));

#pragma unroll
    for (int np = 0; np < 4; ++np) {      // 32-key groups
#pragma unroll
      for (int t2 = 0; t2 < 2; ++t2) {
        const int nt = np * 2 + t2;
        v16h b0 = load_b16(Kh, LDQ, nt * 16, 0, l, hi);
        v16h b1 = load_b16(Kh, LDQ, nt * 16, 32, l, hi);
        v8f s = {0.f, 0.f, 0.f, 0.f, 0.f, 0.f, 0.f, 0.f};
        s = __builtin_amdgcn_wmma_f32_16x16x32_f16(false, a0, false, b0, (short)0, s, false, false);
        s = __builtin_amdgcn_wmma_f32_16x16x32_f16(false, a1, false, b1, (short)0, s, false, false);
        const int word = nt >> 1;
        const int shift = ((nt & 1) << 4) + l;
        const size_t tBase = aBase + (size_t)(w * 16 + 8 * hi) * SEQ +
                             (size_t)c * CK + nt * 16 + l;
#pragma unroll
        for (int v = 0; v < 8; ++v) {
          unsigned bit = (mrA[v][word] >> shift) & 1u;
          float z = bit ? ZMASKED : s[v];
          float p = __builtin_amdgcn_exp2f(z + rowC[v]);  // add + trans only
          __builtin_nontemporal_store(p, attnOut + tBase + (size_t)v * SEQ);
          P16w[(size_t)(v + 8 * hi) * LDP + t2 * 16 + l] = (_Float16)p;
        }
      }
      asm volatile("s_wait_dscnt 0x0" ::: "memory");   // P staging -> A loads
      v16h ap = load_a16(P16w, LDP, 0, 0, l, hi);
#pragma unroll
      for (int dt = 0; dt < 4; ++dt) {
        v16h bv = load_b16(Vt, LDVT, dt * 16, np * 32, l, hi);
        acc[dt] = __builtin_amdgcn_wmma_f32_16x16x32_f16(false, ap, false, bv, (short)0, acc[dt], false, false);
      }
    }
  }

  // ---- write context ----
  const size_t cRow = cBase + (size_t)(w * 16 + 8 * hi) * DK;
#pragma unroll
  for (int dt = 0; dt < 4; ++dt)
#pragma unroll
    for (int v = 0; v < 8; ++v)
      ctxOut[cRow + (size_t)v * DK + dt * 16 + l] = acc[dt][v];
}

extern "C" void kernel_launch(void* const* d_in, const int* in_sizes, int n_in,
                              void* d_out, int out_size, void* d_ws, size_t ws_size,
                              hipStream_t stream) {
  const float*         Q  = (const float*)d_in[0];
  const float*         K  = (const float*)d_in[1];
  const float*         V  = (const float*)d_in[2];
  const unsigned char* Mk = (const unsigned char*)d_in[3];

  float* ctxOut  = (float*)d_out;                         // [4,16,2048,64]
  float* attnOut = ctxOut + (size_t)4 * 16 * 2048 * 64;   // [4,16,2048,2048]

  dim3 grid(4 * 16 * (SEQ / TQ));   // 1024 blocks: (b*h) x query-block
  dim3 block(256);                  // 8 waves of 32
  sdpa_fused_kernel<<<grid, block, 0, stream>>>(Q, K, V, Mk, ctxOut, attnOut);
}
